// BertSelfAttention_14809047236953
// MI455X (gfx1250) — compile-verified, matrix operand order checked by reading, not run on hardware
//
#include <hip/hip_runtime.h>
#include <hip/hip_bf16.h>
#include <stdint.h>

// ---------------------------------------------------------------------------
// Problem constants (BERT self-attention)
// ---------------------------------------------------------------------------
#define BATCH 2
#define SEQ   2048
#define HID   1024
#define NH    16
#define HD    64
#define QK_SCALE 0.125f   // 1/sqrt(64)

// ---------------------------------------------------------------------------
// WMMA fragment types (gfx1250, wave32)
// ---------------------------------------------------------------------------
typedef __attribute__((ext_vector_type(16))) __bf16 v16bf;
typedef __attribute__((ext_vector_type(8)))  float  v8f;

union Frag {            // 16x32 bf16 A-fragment (or 32x16 B-fragment): 8 VGPRs
    v16bf v;
    uint4 q[2];
};
union Acc {             // 16x16 f32 C/D fragment: 8 VGPRs
    v8f   v;
    float f[8];
};

// float -> bf16 bits, round-to-nearest-even
__device__ __forceinline__ uint16_t f2bf(float f) {
    uint32_t u = __float_as_uint(f);
    u += 0x7FFFu + ((u >> 16) & 1u);
    return (uint16_t)(u >> 16);
}

// Load an A/B fragment from a row-major bf16 matrix (global or LDS).
// Layout per ISA 7.12.2 (16-bit A 16x32): lane L holds row (L&15); K chunks
// {0..7,16..23} for lanes 0-15 and {8..15,24..31} for lanes 16-31.
__device__ __forceinline__ Frag load_frag(const uint16_t* base, int ld,
                                          int i0, int k0, int lane) {
    Frag f;
    const int i  = i0 + (lane & 15);
    const int kb = k0 + ((lane >> 4) << 3);
    const uint16_t* p = base + (size_t)i * ld + kb;
    f.q[0] = *reinterpret_cast<const uint4*>(p);        // K = kb .. kb+7
    f.q[1] = *reinterpret_cast<const uint4*>(p + 16);   // K = kb+16 .. kb+23
    return f;
}

__device__ __forceinline__ v8f wmma_bf16(const Frag& a, const Frag& b, v8f c) {
    return __builtin_amdgcn_wmma_f32_16x16x32_bf16(
        /*neg_a=*/false, a.v, /*neg_b=*/false, b.v,
        /*c_mod=*/(short)0, c, /*reuse_a=*/false, /*reuse_b=*/false);
}

// ---------------------------------------------------------------------------
// Async global->LDS staging (CDNA5 GLOBAL_LOAD_ASYNC_TO_LDS_B128, ASYNCcnt).
// Builtin signature (from toolchain diagnostic): b128 payload type is a
// 16-byte int vector; args are (global src, lds dst, imm offset, imm cpol).
// Guarded: falls back to register-bounce staging if the builtin is absent.
// ---------------------------------------------------------------------------
#if __has_builtin(__builtin_amdgcn_global_load_async_to_lds_b128)
#define ASYNC_COPY 1
#define AS1 __attribute__((address_space(1)))
#define AS3 __attribute__((address_space(3)))
typedef int gv4i __attribute__((vector_size(16)));
#endif

__device__ __forceinline__ void copy16B(const uint16_t* g, uint16_t* l) {
#if defined(ASYNC_COPY)
    __builtin_amdgcn_global_load_async_to_lds_b128(
        (AS1 gv4i*)(g), (AS3 gv4i*)(l), 0, 0);
#else
    *reinterpret_cast<uint4*>(l) = *reinterpret_cast<const uint4*>(g);
#endif
}

__device__ __forceinline__ void wait_async_copies() {
#if defined(ASYNC_COPY)
#if __has_builtin(__builtin_amdgcn_s_wait_asynccnt)
    __builtin_amdgcn_s_wait_asynccnt(0);
#else
    asm volatile("s_wait_asynccnt 0x0" ::: "memory");
#endif
#endif
}

// ---------------------------------------------------------------------------
// Phase 0: fp32 -> bf16 staging
// ---------------------------------------------------------------------------
__global__ void cvt_f32_bf16_kernel(const float* __restrict__ in,
                                    uint16_t* __restrict__ out, int n) {
    int i = blockIdx.x * blockDim.x + threadIdx.x;
    int stride = gridDim.x * blockDim.x;
    for (; i < n; i += stride) out[i] = f2bf(in[i]);
}

// ---------------------------------------------------------------------------
// Phase 1: QKV projection.  C = X (BS x H, bf16) * W^T (H x H, bf16) + bias.
// 1x4 register blocking: each wave owns a 16x64 strip (4 N-tiles), loading
// the A fragment once per k-chunk and issuing 4 WMMAs against it (4x less X
// traffic, 4 wmma per A-load).  blockIdx.z selects Q / K / V.
// Q,K stored [B,NH,S,HD]; V stored transposed [B,NH,HD,S] for the ctx GEMM.
// ---------------------------------------------------------------------------
__global__ __launch_bounds__(128)
void qkv_gemm_kernel(const uint16_t* __restrict__ Xb,
                     const uint16_t* __restrict__ Wqb,
                     const uint16_t* __restrict__ Wkb,
                     const uint16_t* __restrict__ Wvb,
                     const float* __restrict__ bq,
                     const float* __restrict__ bk,
                     const float* __restrict__ bv,
                     uint16_t* __restrict__ Qb,
                     uint16_t* __restrict__ Kb,
                     uint16_t* __restrict__ Vt) {
    const int lane = threadIdx.x & 31;
    const int wave = threadIdx.x >> 5;
    const int mt   = blockIdx.x;                     // 0..255  (BS/16 M-tiles)
    const int nt0  = (blockIdx.y * 4 + wave) * 4;    // first of 4 N-tiles
    const int mat  = blockIdx.z;                     // 0=Q 1=K 2=V

    const uint16_t* W    = (mat == 0) ? Wqb : (mat == 1) ? Wkb : Wvb;
    const float*    bias = (mat == 0) ? bq  : (mat == 1) ? bk  : bv;

    Acc c[4] = {};
    for (int k0 = 0; k0 < HID; k0 += 32) {
        Frag a = load_frag(Xb, HID, mt * 16, k0, lane);
        #pragma unroll
        for (int t = 0; t < 4; ++t) {
            Frag b = load_frag(W, HID, (nt0 + t) * 16, k0, lane);
            c[t].v = wmma_bf16(a, b, c[t].v);
        }
    }

    const int col   = lane & 15;
    const int rbase = (lane >> 4) << 3;              // +0 / +8 row split

    #pragma unroll
    for (int t = 0; t < 4; ++t) {
        const int   colN = (nt0 + t) * 16 + col;
        const float bcol = bias[colN];
        const int   h    = colN >> 6;                // head
        const int   d    = colN & 63;                // dim within head
        #pragma unroll
        for (int r = 0; r < 8; ++r) {
            const int rowM = mt * 16 + r + rbase;    // global row in [0, B*S)
            const int bg   = rowM >> 11;             // / SEQ
            const int s    = rowM & (SEQ - 1);
            float val = c[t].f[r] + bcol;
            if (mat == 0) val *= QK_SCALE;
            const uint16_t bits = f2bf(val);
            if (mat < 2) {
                uint16_t* dst = (mat == 0) ? Qb : Kb;
                dst[(((size_t)(bg * NH + h) * SEQ) + s) * HD + d] = bits;
            } else {
                Vt[(((size_t)(bg * NH + h) * HD) + d) * SEQ + s] = bits;
            }
        }
    }
}

// ---------------------------------------------------------------------------
// Phase 2: flash attention.  4 waves/block share one (b,h); K/V tiles for 32
// keys are cooperatively staged into LDS once per block (async global->LDS
// when available), then every wave's WMMA fragments come from LDS.
// P is re-shaped C-layout -> A-layout via a per-wave bf16 LDS region; LDS ops
// from one wave are architecturally in-order (DScnt) so no block barrier.
// ---------------------------------------------------------------------------
#define LDS_LD 40   // bf16 elements per P row (16B-aligned stride)

__global__ __launch_bounds__(128)
void flash_attn_kernel(const uint16_t* __restrict__ Q,
                       const uint16_t* __restrict__ K,
                       const uint16_t* __restrict__ Vt,
                       const float* __restrict__ mask,
                       float* __restrict__ out) {
    __shared__ __align__(16) uint16_t sK[32 * HD];          // 32 keys x 64 d
    __shared__ __align__(16) uint16_t sV[HD * 32];          // 64 d x 32 keys
    __shared__ __align__(16) uint16_t sP[4][16 * LDS_LD];   // per-wave P

    const int tid  = threadIdx.x;
    const int lane = tid & 31;
    const int wave = tid >> 5;
    const int qt   = blockIdx.x * 4 + wave;     // 0..127  (S/16 q-tiles)
    const int h    = blockIdx.y;                // head
    const int b    = blockIdx.z;                // batch

    const size_t headOff = (size_t)(b * NH + h) * SEQ * HD;
    const uint16_t* Qh = Q  + headOff;
    const uint16_t* Kh = K  + headOff;
    const uint16_t* Vh = Vt + headOff;          // [HD][SEQ]
    const float*    mk = mask + (size_t)b * SEQ;

    // Q A-fragments for both 32-wide d-chunks (HD = 64)
    Frag aq0 = load_frag(Qh, HD, qt * 16, 0,  lane);
    Frag aq1 = load_frag(Qh, HD, qt * 16, 32, lane);

    float mrow[8], lrow[8];
    #pragma unroll
    for (int r = 0; r < 8; ++r) { mrow[r] = -3.0e38f; lrow[r] = 0.0f; }
    Acc ctx[4] = {};                            // 16 x 64 f32 context

    uint16_t* P = &sP[wave][0];
    const int col   = lane & 15;
    const int rbase = (lane >> 4) << 3;

    for (int j = 0; j < SEQ; j += 32) {
        // ---- cooperative K/V tile staging (128 threads, 16B chunks) ----
        {
            const uint16_t* gK = Kh + (size_t)j * HD;   // contiguous 4KB
            #pragma unroll
            for (int cch = 0; cch < 2; ++cch) {
                const int e = (tid + cch * 128) * 8;    // element offset
                copy16B(gK + e, &sK[e]);
            }
            #pragma unroll
            for (int cch = 0; cch < 2; ++cch) {
                const int ch = tid + cch * 128;         // 0..255
                const int d  = ch >> 2;
                const int ko = (ch & 3) * 8;
                copy16B(Vh + (size_t)d * SEQ + j + ko, &sV[d * 32 + ko]);
            }
            wait_async_copies();
        }
        __syncthreads();

        // ---- scores for 2 key sub-tiles (fragments from LDS) -----------
        Acc s0 = {}, s1 = {};
        {
            Frag bk = load_frag(sK, HD, 0, 0, lane);
            s0.v = wmma_bf16(aq0, bk, s0.v);
            bk   = load_frag(sK, HD, 0, 32, lane);
            s0.v = wmma_bf16(aq1, bk, s0.v);
            bk   = load_frag(sK, HD, 16, 0, lane);
            s1.v = wmma_bf16(aq0, bk, s1.v);
            bk   = load_frag(sK, HD, 16, 32, lane);
            s1.v = wmma_bf16(aq1, bk, s1.v);
        }
        const float mk0 = mk[j + col];
        const float mk1 = mk[j + 16 + col];

        // ---- online softmax over 32 keys -------------------------------
        #pragma unroll
        for (int r = 0; r < 8; ++r) {
            s0.f[r] += mk0;
            s1.f[r] += mk1;
            float rm = fmaxf(s0.f[r], s1.f[r]);
            rm = fmaxf(rm, __shfl_xor(rm, 1, 16));
            rm = fmaxf(rm, __shfl_xor(rm, 2, 16));
            rm = fmaxf(rm, __shfl_xor(rm, 4, 16));
            rm = fmaxf(rm, __shfl_xor(rm, 8, 16));
            const float mnew  = fmaxf(mrow[r], rm);
            const float alpha = __expf(mrow[r] - mnew);
            s0.f[r] = __expf(s0.f[r] - mnew);
            s1.f[r] = __expf(s1.f[r] - mnew);
            float rs = s0.f[r] + s1.f[r];
            rs += __shfl_xor(rs, 1, 16);
            rs += __shfl_xor(rs, 2, 16);
            rs += __shfl_xor(rs, 4, 16);
            rs += __shfl_xor(rs, 8, 16);
            lrow[r] = lrow[r] * alpha + rs;
            mrow[r] = mnew;
            #pragma unroll
            for (int dt = 0; dt < 4; ++dt) ctx[dt].f[r] *= alpha;
        }

        // ---- C-layout P -> bf16 LDS -> A-layout fragment (per-wave) ----
        #pragma unroll
        for (int r = 0; r < 8; ++r) {
            const int row = r + rbase;
            P[row * LDS_LD + col]      = f2bf(s0.f[r]);
            P[row * LDS_LD + 16 + col] = f2bf(s1.f[r]);
        }
        __builtin_amdgcn_wave_barrier();   // same-wave LDS is in-order (DScnt)
        Frag ap;
        {
            const int arow = lane & 15;
            const int ks   = (lane >> 4) << 3;
            ap.q[0] = *reinterpret_cast<const uint4*>(&P[arow * LDS_LD + ks]);
            ap.q[1] = *reinterpret_cast<const uint4*>(&P[arow * LDS_LD + ks + 16]);
        }

        // ---- ctx += P x V^T (V fragments from LDS) ---------------------
        #pragma unroll
        for (int dt = 0; dt < 4; ++dt) {
            Frag bv = load_frag(sV, 32, dt * 16, 0, lane);
            ctx[dt].v = wmma_bf16(ap, bv, ctx[dt].v);
        }
        __syncthreads();   // all waves done with sK/sV before restaging
    }

    // ---- normalize + store [B,S,H] fp32 --------------------------------
    #pragma unroll
    for (int dt = 0; dt < 4; ++dt) {
        #pragma unroll
        for (int r = 0; r < 8; ++r) {
            const int rowS = qt * 16 + r + rbase;
            const int colH = h * HD + dt * 16 + col;
            out[((size_t)(b * SEQ + rowS)) * HID + colH] = ctx[dt].f[r] / lrow[r];
        }
    }
}

// ---------------------------------------------------------------------------
// Launch
// ---------------------------------------------------------------------------
extern "C" void kernel_launch(void* const* d_in, const int* in_sizes, int n_in,
                              void* d_out, int out_size, void* d_ws, size_t ws_size,
                              hipStream_t stream) {
    const float* hidden = (const float*)d_in[0];
    const float* amask  = (const float*)d_in[1];
    const float* Wq     = (const float*)d_in[2];
    const float* bq     = (const float*)d_in[3];
    const float* Wk     = (const float*)d_in[4];
    const float* bk     = (const float*)d_in[5];
    const float* Wv     = (const float*)d_in[6];
    const float* bv     = (const float*)d_in[7];
    float* out = (float*)d_out;

    const size_t nX = (size_t)BATCH * SEQ * HID;   // 4,194,304
    const size_t nW = (size_t)HID * HID;           // 1,048,576

    char* ws = (char*)d_ws;
    uint16_t* Xb  = (uint16_t*)ws;              ws += nX * 2;
    uint16_t* Wqb = (uint16_t*)ws;              ws += nW * 2;
    uint16_t* Wkb = (uint16_t*)ws;              ws += nW * 2;
    uint16_t* Wvb = (uint16_t*)ws;              ws += nW * 2;
    uint16_t* Qb  = (uint16_t*)ws;              ws += nX * 2;
    uint16_t* Kb  = (uint16_t*)ws;              ws += nX * 2;
    uint16_t* Vt  = (uint16_t*)ws;              ws += nX * 2;

    // Phase 0: stage bf16 copies
    cvt_f32_bf16_kernel<<<512, 256, 0, stream>>>(hidden, Xb, (int)nX);
    cvt_f32_bf16_kernel<<<512, 256, 0, stream>>>(Wq, Wqb, (int)nW);
    cvt_f32_bf16_kernel<<<512, 256, 0, stream>>>(Wk, Wkb, (int)nW);
    cvt_f32_bf16_kernel<<<512, 256, 0, stream>>>(Wv, Wvb, (int)nW);

    // Phase 1: QKV projections (256 M-tiles x 4 N-strip groups x {Q,K,V})
    qkv_gemm_kernel<<<dim3((BATCH * SEQ) / 16, (HID / 16) / 16, 3), 128, 0, stream>>>(
        Xb, Wqb, Wkb, Wvb, bq, bk, bv, Qb, Kb, Vt);

    // Phase 2: flash attention (32 q-tile groups x 16 heads x 2 batch)
    flash_attn_kernel<<<dim3((SEQ / 16) / 4, NH, BATCH), 128, 0, stream>>>(
        Qb, Kb, Vt, amask, out);

    (void)in_sizes; (void)n_in; (void)out_size; (void)ws_size;
}
